// ChannelTimeAttention_3607772528801
// MI455X (gfx1250) — matrix operand order repaired and checked by reading, not compiled, not run on hardware
//
#include <hip/hip_runtime.h>
#include <hip/hip_bf16.h>

#define BDIM 16
#define CDIM 256
#define LDIM 2048
#define NTOT ((long)BDIM * CDIM * LDIM)   // 8,388,608 elements

typedef __bf16 bf16_t;
typedef __attribute__((ext_vector_type(8)))  __bf16 v8bf;
typedef __attribute__((ext_vector_type(16))) __bf16 v16bf;
typedef __attribute__((ext_vector_type(8)))  float  v8f;
typedef __attribute__((ext_vector_type(4)))  float  f32x4;

union V16U { v16bf v; v8bf h[2]; };
union PBU  { v16bf v; unsigned u[8]; };
union PKU  { unsigned u; __bf16 h[2]; };

__device__ inline v8f wmma_bf16(v16bf a, v16bf b, v8f c) {
  // D = A(16x32 bf16) x B(32x16 bf16) + C(16x16 f32)
  return __builtin_amdgcn_wmma_f32_16x16x32_bf16(false, a, false, b,
                                                 (short)0, c, false, false);
}

// A-operand 16x32 tile, row-major source. Caller passes per-lane pointer:
//   p = rowbase(row = lane&15) + 16*(lane>>4)  ; loads 16 contiguous
__device__ inline v16bf load_a16x32(const bf16_t* p) {
  V16U u;
  u.h[0] = *(const v8bf*)p;
  u.h[1] = *(const v8bf*)(p + 8);
  return u.v;
}

// B-operand 32x16 tile from B^T rows. Caller passes per-lane pointer:
//   p = BT_rowbase(col = lane&15) + 16*(lane>>4) ; loads 16 contiguous
__device__ inline v16bf load_b32x16(const bf16_t* p) {
  V16U u;
  u.h[0] = *(const v8bf*)p;
  u.h[1] = *(const v8bf*)(p + 8);
  return u.v;
}

__device__ inline unsigned pack2bf(float a, float b) {
  PKU r; r.h[0] = (__bf16)a; r.h[1] = (__bf16)b; return r.u;
}

// ---------------------------------------------------------------------------
// Pass 1: f32 -> bf16 convert + build transposed copy  yT[B,L,C]
// ---------------------------------------------------------------------------
__global__ __launch_bounds__(256) void k_convert(const float* __restrict__ y,
                                                 bf16_t* __restrict__ ybf,
                                                 bf16_t* __restrict__ ytbf) {
  __shared__ __bf16 tile[32][33];
  const int b  = blockIdx.z;
  const int c0 = blockIdx.y * 32;
  const int l0 = blockIdx.x * 32;
  const int tx = threadIdx.x, ty = threadIdx.y;
#pragma unroll
  for (int i = 0; i < 32; i += 8) {
    const int c = c0 + ty + i;
    const long idx = ((long)(b * CDIM + c)) * LDIM + l0 + tx;
    const bf16_t h = (bf16_t)y[idx];
    ybf[idx] = h;
    tile[tx][ty + i] = h;        // tile[l_local][c_local]
  }
  __syncthreads();
#pragma unroll
  for (int i = 0; i < 32; i += 8) {
    const int l = l0 + ty + i;
    ytbf[((long)(b * LDIM + l)) * CDIM + c0 + tx] = tile[ty + i][tx];
  }
}

// ---------------------------------------------------------------------------
// Pass 2: channel scores  S_c[b] = (y y^T) * 1/sqrt(L)   [B,256,256] f32
// one wave per 16x16 tile, K = L = 2048
// ---------------------------------------------------------------------------
__global__ __launch_bounds__(32) void k_scores_c(const bf16_t* __restrict__ ybf,
                                                 float* __restrict__ sc) {
  const int lane = threadIdx.x, hi = lane >> 4, lr = lane & 15;
  const int b = blockIdx.z, i0 = blockIdx.y * 16, j0 = blockIdx.x * 16;
  const bf16_t* aBase = ybf + ((long)(b * CDIM + i0 + lr)) * LDIM + (hi << 4);
  const bf16_t* bBase = ybf + ((long)(b * CDIM + j0 + lr)) * LDIM + (hi << 4);
  v8f acc = {};
#pragma unroll 4
  for (int k0 = 0; k0 < LDIM; k0 += 32)
    acc = wmma_bf16(load_a16x32(aBase + k0), load_b32x16(bBase + k0), acc);
  const float scale = 0.022097086912079608f;  // 1/sqrt(2048)
  float* orow = sc + ((long)(b * CDIM + i0 + (hi << 3))) * CDIM + j0 + lr;
#pragma unroll
  for (int r = 0; r < 8; ++r) orow[(long)r * CDIM] = acc[r] * scale;
}

// ---------------------------------------------------------------------------
// Pass 3: row softmax of S_c -> bf16 attn. One wave per 256-wide row.
// ---------------------------------------------------------------------------
__global__ __launch_bounds__(32) void k_softmax(const float* __restrict__ sc,
                                                bf16_t* __restrict__ attn) {
  const long row = blockIdx.x;
  const int lane = threadIdx.x;
  const float* p = sc + row * CDIM + lane * 8;
  f32x4 a0 = *(const f32x4*)p;
  f32x4 a1 = *(const f32x4*)(p + 4);
  float v[8];
#pragma unroll
  for (int j = 0; j < 4; ++j) { v[j] = a0[j]; v[j + 4] = a1[j]; }
  float m = -1e30f;
#pragma unroll
  for (int j = 0; j < 8; ++j) m = fmaxf(m, v[j]);
  for (int off = 1; off < 32; off <<= 1) m = fmaxf(m, __shfl_xor(m, off));
  float s = 0.f;
#pragma unroll
  for (int j = 0; j < 8; ++j) { v[j] = __expf(v[j] - m); s += v[j]; }
  for (int off = 1; off < 32; off <<= 1) s += __shfl_xor(s, off);
  const float inv = 1.f / s;
  v8bf o;
#pragma unroll
  for (int j = 0; j < 8; ++j) o[j] = (bf16_t)(v[j] * inv);
  *(v8bf*)(attn + row * CDIM + lane * 8) = o;
}

// ---------------------------------------------------------------------------
// Pass 4: y_c = attn (C x C) @ y (C x L)  -> f32 [B,C,L]
// B^T rows come from yT_bf (contiguous). One wave per 16x16 tile, K=256.
// ---------------------------------------------------------------------------
__global__ __launch_bounds__(32) void k_yc(const bf16_t* __restrict__ attn,
                                           const bf16_t* __restrict__ ytbf,
                                           float* __restrict__ yc) {
  const int lane = threadIdx.x, hi = lane >> 4, lr = lane & 15;
  const int b = blockIdx.z, i0 = blockIdx.y * 16, j0 = blockIdx.x * 16;
  const bf16_t* aBase = attn + ((long)(b * CDIM + i0 + lr)) * CDIM + (hi << 4);
  const bf16_t* bBase = ytbf + ((long)(b * LDIM + j0 + lr)) * CDIM + (hi << 4);
  v8f acc = {};
#pragma unroll
  for (int k0 = 0; k0 < CDIM; k0 += 32)
    acc = wmma_bf16(load_a16x32(aBase + k0), load_b32x16(bBase + k0), acc);
  float* orow = yc + ((long)(b * CDIM + i0 + (hi << 3))) * LDIM + j0 + lr;
#pragma unroll
  for (int r = 0; r < 8; ++r) orow[(long)r * LDIM] = acc[r];
}

// ---------------------------------------------------------------------------
// Pass 5: flash time-attention (computed transposed) + final combine.
// One wave handles 16 query rows (l0..l0+15) x 128 channels (c0..c0+127),
// streaming keys/values in chunks of 32 along m.
//   S^T = K_tile(16m x C) @ Q^T     -> lane holds fixed q = lane&15
//   O^T = V^T(16c x 32m) @ P^T      -> per-lane softmax corrections
// out = alpha*y_c + beta*O^T/l + gamma*y
// ---------------------------------------------------------------------------
__global__ __launch_bounds__(32) void k_flash(const bf16_t* __restrict__ ybf,
                                              const bf16_t* __restrict__ ytbf,
                                              const float* __restrict__ y,
                                              const float* __restrict__ yc,
                                              const float* __restrict__ pA,
                                              const float* __restrict__ pB,
                                              const float* __restrict__ pG,
                                              float* __restrict__ out) {
  const int lane = threadIdx.x, hi = lane >> 4, lr = lane & 15;
  const int b  = blockIdx.z;
  const int l0 = blockIdx.y * 16;
  const int c0 = blockIdx.x * 128;

  // Resident queries as B-operand tiles (K = channel dim, 8 chunks of 32)
  v16bf Qb[8];
  {
    const bf16_t* qrow = ytbf + ((long)(b * LDIM + l0 + lr)) * CDIM + (hi << 4);
#pragma unroll
    for (int kc = 0; kc < 8; ++kc) Qb[kc] = load_b32x16(qrow + 32 * kc);
  }

  v8f O[8];
#pragma unroll
  for (int ct = 0; ct < 8; ++ct) O[ct] = (v8f){};
  float rmax = -1e30f, rsum = 0.f;

  for (int m0 = 0; m0 < LDIM; m0 += 32) {
    // --- S^T tiles for keys m0..m0+15 and m0+16..m0+31 (K = C = 256) ---
    v8f s0 = {}, s1 = {};
    const bf16_t* kBase = ytbf + ((long)(b * LDIM + m0 + lr)) * CDIM + (hi << 4);
#pragma unroll
    for (int kc = 0; kc < 8; ++kc) {
      s0 = wmma_bf16(load_a16x32(kBase + 32 * kc), Qb[kc], s0);
      s1 = wmma_bf16(load_a16x32(kBase + (long)16 * CDIM + 32 * kc), Qb[kc], s1);
    }
    // --- online softmax over m (per-lane: q = lane&15) ---
    const float sct = 0.0625f;  // 1/sqrt(256)
    float lm = -1e30f;
#pragma unroll
    for (int r = 0; r < 8; ++r) {
      s0[r] *= sct; s1[r] *= sct;
      lm = fmaxf(lm, fmaxf(s0[r], s1[r]));
    }
    lm = fmaxf(lm, __shfl_xor(lm, 16));
    const float nm = fmaxf(rmax, lm);
    const float corr = __expf(rmax - nm);
    float p0[8], p1[8], ps = 0.f;
#pragma unroll
    for (int r = 0; r < 8; ++r) {
      p0[r] = __expf(s0[r] - nm);
      p1[r] = __expf(s1[r] - nm);
      ps += p0[r] + p1[r];
    }
    ps += __shfl_xor(ps, 16);
    rsum = rsum * corr + ps;
    rmax = nm;
#pragma unroll
    for (int ct = 0; ct < 8; ++ct)
#pragma unroll
      for (int r = 0; r < 8; ++r) O[ct][r] *= corr;

    // --- repack P into B-operand (P^T) layout: 8 xor-16 shuffles ---
    unsigned x[4], yv[4], sx[4], sy[4];
#pragma unroll
    for (int j = 0; j < 4; ++j) {
      x[j]  = pack2bf(p0[2 * j], p0[2 * j + 1]);
      yv[j] = pack2bf(p1[2 * j], p1[2 * j + 1]);
    }
#pragma unroll
    for (int j = 0; j < 4; ++j) {
      sx[j] = __shfl_xor(x[j], 16);
      sy[j] = __shfl_xor(yv[j], 16);
    }
    PBU Pb;
#pragma unroll
    for (int j = 0; j < 4; ++j) {
      Pb.u[j]     = hi ? sy[j] : x[j];   // K 0..7  (hi=0) / 16..23 (hi=1)
      Pb.u[4 + j] = hi ? yv[j] : sx[j];  // K 8..15 (hi=0) / 24..31 (hi=1)
    }

    // --- O^T accumulation: A tiles straight from row-major y_bf [C,L] ---
    const bf16_t* vBase = ybf + ((long)(b * CDIM + c0 + lr)) * LDIM + m0 + (hi << 4);
#pragma unroll
    for (int ct = 0; ct < 8; ++ct)
      O[ct] = wmma_bf16(load_a16x32(vBase + (long)(16 * ct) * LDIM), Pb.v, O[ct]);
  }

  // --- epilogue: normalize + combine + store ---
  const float alpha = pA[0], beta = pB[0], gamma = pG[0];
  const float inv = 1.f / rsum;
#pragma unroll
  for (int ct = 0; ct < 8; ++ct) {
#pragma unroll
    for (int r = 0; r < 8; ++r) {
      const int c = c0 + 16 * ct + r + (hi << 3);
      const long idx = ((long)(b * CDIM + c)) * LDIM + l0 + lr;
      out[idx] = alpha * yc[idx] + beta * (O[ct][r] * inv) + gamma * y[idx];
    }
  }
}

// ---------------------------------------------------------------------------
extern "C" void kernel_launch(void* const* d_in, const int* in_sizes, int n_in,
                              void* d_out, int out_size, void* d_ws, size_t ws_size,
                              hipStream_t stream) {
  const float* y     = (const float*)d_in[0];
  const float* alpha = (const float*)d_in[1];
  const float* beta  = (const float*)d_in[2];
  const float* gamma = (const float*)d_in[3];
  float* out = (float*)d_out;

  char* ws = (char*)d_ws;
  bf16_t* ybf  = (bf16_t*)ws;                                   // 16 MB
  bf16_t* ytbf = ybf + NTOT;                                    // 16 MB
  float*  sc   = (float*)(ws + 2 * NTOT * sizeof(bf16_t));      //  4 MB
  bf16_t* attn = (bf16_t*)((char*)sc + (long)BDIM * CDIM * CDIM * sizeof(float)); // 2 MB
  float*  yc   = (float*)((char*)attn + (long)BDIM * CDIM * CDIM * sizeof(bf16_t)); // 32 MB

  k_convert<<<dim3(LDIM / 32, CDIM / 32, BDIM), dim3(32, 8), 0, stream>>>(y, ybf, ytbf);
  k_scores_c<<<dim3(CDIM / 16, CDIM / 16, BDIM), 32, 0, stream>>>(ybf, sc);
  k_softmax<<<dim3(BDIM * CDIM), 32, 0, stream>>>(sc, attn);
  k_yc<<<dim3(LDIM / 16, CDIM / 16, BDIM), 32, 0, stream>>>(attn, ytbf, yc);
  k_flash<<<dim3(2, LDIM / 16, BDIM), 32, 0, stream>>>(ybf, ytbf, y, yc,
                                                       alpha, beta, gamma, out);
}
